// MLP_1314259992962
// MI455X (gfx1250) — compile-verified, matrix operand order checked by reading, not run on hardware
//
#include <hip/hip_runtime.h>

// ---------------------------------------------------------------------------
// MLP chain: 16 x (Linear(128,128) + ReLU) then Linear(128,2), N = 1M rows.
// Compute-bound (~1000 FLOP/byte) -> maximize v_wmma_f32_16x16x32_f16 density.
//
// Orientation: D = A x B with A = weight tile (M=out-neuron j, K=in), and
// B = activations with the BATCH on the N axis. Then:
//   D layout  : lane t -> batch row t%16, j = r + 8*(t/16) + 16*jt
//   B layout  : (mirror-of-A striping, ISA 7.12.2 "simplified layout")
//               lane t -> batch row t%16, K = 2*(v%4) + 16*(v/4) + 8*(t/16)
// With j==K(next layer), the 8*(t/16) terms cancel and the D->B relayout is
// PURELY per-lane:  B[kt][v] = pk_max(cvt_pk_f16(acc[2kt+v/4][2(v%4)..+1]),0).
// No LDS, no barriers between layers; activations live in VGPRs for the whole
// 16-layer chain. Weight A-fragments stream through a depth-2 software
// pipeline (2 fragment loads always in flight, crossing layer boundaries).
// ---------------------------------------------------------------------------

typedef _Float16 v8h  __attribute__((ext_vector_type(8)));
typedef _Float16 v16h __attribute__((ext_vector_type(16)));
typedef float    v8f  __attribute__((ext_vector_type(8)));
typedef float    v4f  __attribute__((ext_vector_type(4)));

#define NLAYER 16
#define HDIM   128

// One-time fp32 -> f16 weight conversion into workspace (same [l][j][k] layout).
__global__ void convert_w_f16(const float* __restrict__ W,
                              _Float16* __restrict__ W16, int n) {
  int i = blockIdx.x * blockDim.x + threadIdx.x;
  if (i < n) W16[i] = (_Float16)W[i];
}

// Load weight A-fragment i (i = jt*4 + kt) of one layer.
// A element (g*4+m)*2+e <-> K = 32kt + 16g + 8hi + 2m + e, row j = 16jt+l15.
__device__ __forceinline__ v16h loadA(const _Float16* base, int i, int l15, int hi) {
  const _Float16* pa = base + (((i >> 2) * 16 + l15) * HDIM) + (i & 3) * 32 + hi * 8;
  v8h lo8 = *(const v8h*)(pa);
  v8h hi8 = *(const v8h*)(pa + 16);
  v16h A;
#pragma unroll
  for (int j = 0; j < 8; ++j) { A[j] = lo8[j]; A[8 + j] = hi8[j]; }
  return A;
}

// One layer: acc = bias; 64 WMMAs streaming A fragments through the 2-slot
// pipeline Af. If PRELOAD_NEXT, fragments 0/1 of the next layer are issued at
// i==30/31 so they fly underneath the caller's relayout VALU.
template <bool PRELOAD_NEXT>
__device__ __forceinline__ void layer_step(const _Float16* wl,
                                           const _Float16* wnext,
                                           const float* bl,
                                           v16h (&B)[2][4], v8f (&acc)[2][8],
                                           v16h (&Af)[2], int l15, int hi) {
  // acc = bias (broadcast over batch rows; acc elem r <-> j = 16jt + 8hi + r)
#pragma unroll
  for (int jt = 0; jt < 8; ++jt) {
    const float* pb = bl + jt * 16 + hi * 8;
    v4f b0 = *(const v4f*)(pb);
    v4f b1 = *(const v4f*)(pb + 4);
#pragma unroll
    for (int j = 0; j < 4; ++j) {
      acc[0][jt][j]     = b0[j];  acc[0][jt][4 + j] = b1[j];
      acc[1][jt][j]     = b0[j];  acc[1][jt][4 + j] = b1[j];
    }
  }

#pragma unroll
  for (int i = 0; i < 32; ++i) {
    const int jt = i >> 2, kt = i & 3;
    v16h Acur = Af[i & 1];
    if (i + 2 < 32) {
      Af[i & 1] = loadA(wl, i + 2, l15, hi);
    } else if (PRELOAD_NEXT) {
      Af[i & 1] = loadA(wnext, i + 2 - 32, l15, hi);
    }
    acc[0][jt] = __builtin_amdgcn_wmma_f32_16x16x32_f16(
        false, Acur, false, B[0][kt], (short)0, acc[0][jt], false, false);
    acc[1][jt] = __builtin_amdgcn_wmma_f32_16x16x32_f16(
        false, Acur, false, B[1][kt], (short)0, acc[1][jt], false, false);
  }
}

__launch_bounds__(128, 1)
__global__ void mlp_wmma(const float* __restrict__ x,
                         const _Float16* __restrict__ W16,
                         const float* __restrict__ b,
                         const float* __restrict__ Wout,
                         const float* __restrict__ bout,
                         float* __restrict__ out,
                         int N) {
  const int lane = threadIdx.x & 31;
  const int wave = threadIdx.x >> 5;
  const int hi   = lane >> 4;   // half-wave select (K/M +8 offset)
  const int l15  = lane & 15;   // batch row within 16-tile / weight row
  const long rowbase = ((long)blockIdx.x * 4 + wave) * 32;  // 32 rows per wave
  if (rowbase >= N) return;

  v16h B[2][4];   // activations: 2 batch sets x 4 K-fragments (16x16x32 B op)
  v8f  acc[2][8]; // f32 accumulators: 2 batch sets x 8 output-neuron tiles
  v16h Af[2];     // weight A-fragment pipeline (depth 2)

  const v16h zero16 = (v16h)(_Float16)0.0f;

  // ---- Load x (fp32) -> initial B fragments (f16), per-lane gather.
  // B element (g*4+m)*2+e  <->  K = 32*kt + 16*g + 8*hi + 2m + e
#pragma unroll
  for (int s = 0; s < 2; ++s) {
    const float* px = x + (rowbase + s * 16 + l15) * HDIM + hi * 8;
#pragma unroll
    for (int kt = 0; kt < 4; ++kt) {
#pragma unroll
      for (int g = 0; g < 2; ++g) {
        const float* p = px + kt * 32 + g * 16;
        v4f f0 = *(const v4f*)(p);
        v4f f1 = *(const v4f*)(p + 4);
#pragma unroll
        for (int j = 0; j < 4; ++j) {
          B[s][kt][g * 8 + j]     = (_Float16)f0[j];
          B[s][kt][g * 8 + 4 + j] = (_Float16)f1[j];
        }
      }
    }
  }

  const _Float16* wl = W16;
  const float*    bl = b;

  // Prime the A-fragment pipeline with layer 0, fragments 0 and 1.
  Af[0] = loadA(wl, 0, l15, hi);
  Af[1] = loadA(wl, 1, l15, hi);

#pragma unroll 1
  for (int l = 0; l < NLAYER - 1; ++l) {
    // Warm next layer's weights in WGP$/L2 (256 cachelines of 128B = 32KB).
    {
      const _Float16* nx = wl + HDIM * HDIM;
#pragma unroll
      for (int c = 0; c < 8; ++c)
        __builtin_prefetch(nx + (c * 32 + lane) * 64, 0, 0);
    }

    layer_step<true>(wl, wl + HDIM * HDIM, bl, B, acc, Af, l15, hi);

    // ---- Per-lane relayout: D tiles -> next layer's B fragments.
    // cvt_pk_f16_f32 pairs, then packed ReLU (v_pk_max_num_f16); for RNE f16,
    // max(cvt(x),0) == cvt(max(x,0)), so this matches relu-then-convert.
    // Runs while the next layer's first two fragment loads are in flight.
#pragma unroll
    for (int s = 0; s < 2; ++s)
#pragma unroll
      for (int kt = 0; kt < 4; ++kt) {
        v16h t;
#pragma unroll
        for (int g = 0; g < 2; ++g)
#pragma unroll
          for (int m = 0; m < 4; ++m) {
            t[(g * 4 + m) * 2]     = (_Float16)acc[s][2 * kt + g][2 * m];
            t[(g * 4 + m) * 2 + 1] = (_Float16)acc[s][2 * kt + g][2 * m + 1];
          }
        B[s][kt] = __builtin_elementwise_max(t, zero16);
      }

    wl += HDIM * HDIM;
    bl += HDIM;
  }

  // Final hidden layer (no next-layer preload, no relayout).
  layer_step<false>(wl, wl, bl, B, acc, Af, l15, hi);

  // ---- Output head: out[r][o] = sum_j relu(h[r][j]) * Wout[o][j] + bout[o]
  const float bo0 = bout[0], bo1 = bout[1];
#pragma unroll
  for (int s = 0; s < 2; ++s) {
    float s0 = 0.0f, s1 = 0.0f;
#pragma unroll
    for (int jt = 0; jt < 8; ++jt) {
      const float* p0 = Wout + jt * 16 + hi * 8;           // o = 0
      const float* p1 = Wout + HDIM + jt * 16 + hi * 8;    // o = 1
      v4f a0 = *(const v4f*)(p0), a1 = *(const v4f*)(p0 + 4);
      v4f c0 = *(const v4f*)(p1), c1 = *(const v4f*)(p1 + 4);
#pragma unroll
      for (int j = 0; j < 4; ++j) {
        float h0 = fmaxf(acc[s][jt][j],     0.0f);
        float h1 = fmaxf(acc[s][jt][4 + j], 0.0f);
        s0 = fmaf(h0, a0[j], s0);  s0 = fmaf(h1, a1[j], s0);
        s1 = fmaf(h0, c0[j], s1);  s1 = fmaf(h1, c1[j], s1);
      }
    }
    // lanes t and t+16 hold complementary j-halves of the same batch row
    s0 += __shfl_xor(s0, 16, 32);
    s1 += __shfl_xor(s1, 16, 32);
    if (lane < 16) {
      long row = rowbase + s * 16 + l15;
      reinterpret_cast<float2*>(out)[row] = make_float2(s0 + bo0, s1 + bo1);
    }
  }
}

extern "C" void kernel_launch(void* const* d_in, const int* in_sizes, int n_in,
                              void* d_out, int out_size, void* d_ws, size_t ws_size,
                              hipStream_t stream) {
  const float* x    = (const float*)d_in[0];
  const float* W    = (const float*)d_in[1];
  const float* b    = (const float*)d_in[2];
  const float* Wout = (const float*)d_in[3];
  const float* bout = (const float*)d_in[4];
  float* out = (float*)d_out;

  const int N  = in_sizes[0] / HDIM;      // 1,048,576 rows
  const int nw = NLAYER * HDIM * HDIM;    // 262,144 weights -> 512KB f16 in d_ws

  _Float16* W16 = (_Float16*)d_ws;
  convert_w_f16<<<(nw + 255) / 256, 256, 0, stream>>>(W, W16, nw);
  // 128 threads = 4 waves (one per SIMD of a WGP), 32 rows/wave -> 128 rows/block
  mlp_wmma<<<N / 128, 128, 0, stream>>>(x, W16, b, Wout, bout, out, N);
}
